// VQ_VAE_5695126634992
// MI455X (gfx1250) — matrix operand order repaired and checked by reading, not compiled, not run on hardware
//
#include <hip/hip_runtime.h>
#include <hip/hip_bf16.h>
#include <hip/hip_fp16.h>

// ---------------------------------------------------------------------------
// VQ-VAE fused forward for MI455X (gfx1250): wave32, WMMA f16->f32,
// TDM tensor_load_to_lds staging, fragment-packed weights streamed from L2.
// ---------------------------------------------------------------------------

typedef __attribute__((ext_vector_type(16))) _Float16 v16h;
typedef __attribute__((ext_vector_type(8)))  float    v8f;
typedef unsigned int u32x4 __attribute__((ext_vector_type(4)));
typedef int          i32x4 __attribute__((ext_vector_type(4)));
typedef int          i32x8 __attribute__((ext_vector_type(8)));

#define B_ROWS   131072
#define FEATURE  514
#define HID      512
#define LATENT   128
#define NUM_EMB  512

#define TILE_M   64
#define THREADS  256          // 8 wave32 waves: 4 m-tiles x 2 n-halves

// padded K/N so every GEMM is whole 16x16x32 tiles
#define KP1      544          // FEATURE -> 17 k-tiles of 32
#define NT4      33           // FEATURE -> 33 n-tiles of 16 (decoder out)

// LDS strides in halves; row stride (dwords) == 4*odd mod 64 so the 16 lanes
// of an A-fragment b128 read hit distinct bank quads.
#define SA_STRIDE 552         // >= 544 ; 276 dw
#define SH_STRIDE 520         // >= 512 ; 260 dw
#define SZ_STRIDE 136         // >= 128 ;  68 dw

// LDS layout (bytes). The fp32 TDM staging buffer is aliased with sH/sZ/sZq:
// stage is dead once sA (f16) is built; sH is first written after that.
#define SA_OFF    0
#define U_OFF     (SA_OFF + TILE_M*SA_STRIDE*2)        // 70656
#define STAGE_OFF U_OFF                                 // 64*514*4 = 131584
#define SH_OFF    U_OFF                                 // 64*520*2 =  66560
#define SZ_OFF    (SH_OFF + TILE_M*SH_STRIDE*2)         // +17408
#define SZQ_OFF   (SZ_OFF + TILE_M*SZ_STRIDE*2)         // +17408
#define REDD_OFF  (U_OFF + TILE_M*FEATURE*4)            // 202240
#define REDI_OFF  (REDD_OFF + TILE_M*2*4)               // 202752
#define IDX_OFF   (REDI_OFF + TILE_M*2*4)               // 203264
#define SMEM_BYTES (IDX_OFF + TILE_M*4)                 // 203520

// ---------------------------------------------------------------------------
// A fragment (16x32 f16, ISA layout): lane l, row = m0+(l&15);
//   halves 0..7  = K[k0 + 8*(l>>4) + 0..7], halves 8..15 = +16.
__device__ inline v16h load_afrag(const _Float16* __restrict__ s, int stride,
                                  int m0, int k0, int lane) {
  const int row = m0 + (lane & 15);
  const int kb  = k0 + ((lane >> 4) << 3);
  const _Float16* p = s + row * stride + kb;
  v16h r;
  uint4* rp = reinterpret_cast<uint4*>(&r);
  rp[0] = *reinterpret_cast<const uint4*>(p);
  rp[1] = *reinterpret_cast<const uint4*>(p + 16);
  return r;
}

// B fragment: pre-packed, fragment f = 512 consecutive halves, 16 per lane.
__device__ inline v16h load_bfrag(const _Float16* __restrict__ wpk,
                                  int fragIdx, int lane) {
  const _Float16* p = wpk + ((size_t)fragIdx << 9) + (lane << 4);
  v16h r;
  uint4* rp = reinterpret_cast<uint4*>(&r);
  rp[0] = *reinterpret_cast<const uint4*>(p);
  rp[1] = *reinterpret_cast<const uint4*>(p + 8);
  return r;
}

__device__ inline v8f vzero() {
  v8f z;
#pragma unroll
  for (int i = 0; i < 8; ++i) z[i] = 0.0f;
  return z;
}

// outer K loop (runtime), inner N tiles unrolled: accumulators stay in VGPRs.
template <int NT>
__device__ inline void gemm_tile(const _Float16* __restrict__ sA, int strideA,
                                 int m0, const _Float16* __restrict__ wpk,
                                 int ktiles, int ntilesTotal, int ntBase,
                                 int lane, v8f* acc) {
  for (int kt = 0; kt < ktiles; ++kt) {
    v16h a = load_afrag(sA, strideA, m0, kt * 32, lane);
#pragma unroll
    for (int j = 0; j < NT; ++j) {
      v16h b = load_bfrag(wpk, kt * ntilesTotal + ntBase + j, lane);
      acc[j] = __builtin_amdgcn_wmma_f32_16x16x32_f16(
          false, a, false, b, (short)0, acc[j], false, false);
    }
  }
}

__device__ inline float elu_f(float x)  { return x > 0.0f ? x : (__expf(x) - 1.0f); }
__device__ inline float sigm_f(float x) { return 1.0f / (1.0f + __expf(-x)); }

// sigmoid epilogue for GEMM4 (n-tiles [ntBase, ntBase+NT))
template <int NT>
__device__ inline void gemm4_epilogue(const v8f* acc, int ntBase, int l15, int rhi,
                                      int m0, int row0, const float* __restrict__ db2,
                                      float* __restrict__ out_xr) {
#pragma unroll
  for (int j = 0; j < NT; ++j) {
    int col = (ntBase + j) * 16 + l15;
    if (col < FEATURE) {
      float bb = db2[col];
#pragma unroll
      for (int i = 0; i < 8; ++i) {
        int r = m0 + i + rhi;
        out_xr[(size_t)(row0 + r) * FEATURE + col] = sigm_f(acc[j][i] + bb);
      }
    }
  }
}

// TDM: DMA rows [row0, row0+64) of x (fp32, row length 514) into LDS stage.
__device__ inline void tdm_load_xtile(const float* __restrict__ x, int row0,
                                      unsigned ldsOff) {
  unsigned long long ga =
      (unsigned long long)(uintptr_t)(x + (size_t)row0 * FEATURE);
  u32x4 g0;
  g0[0] = 1u;                                   // count=1, user descriptor
  g0[1] = ldsOff;                               // lds_addr (bytes)
  g0[2] = (unsigned)(ga & 0xffffffffu);         // global_addr[31:0]
  g0[3] = (unsigned)((ga >> 32) & 0x1ffffffu)   // global_addr[56:32]
          | (2u << 30);                         // type = 2 ("image")
  i32x8 g1;
  g1[0] = (int)(2u << 16);                      // data_size = 4B; no flags
  g1[1] = (int)((unsigned)(FEATURE & 0xffff) << 16);  // tensor_dim0 lo16 @bit48
  g1[2] = 0;                                    // tensor_dim0 hi / dim1 lo16
  g1[3] = (int)(((unsigned)FEATURE << 16) |     // tile_dim0 = 514 @[127:112]
                (unsigned)(B_ROWS >> 16));      // tensor_dim1 hi16
  g1[4] = TILE_M;                               // tile_dim1 = 64; tile_dim2 = 0
  g1[5] = FEATURE;                              // tensor_dim0_stride = 514
  g1[6] = 0;
  g1[7] = 0;
  i32x4 z4; z4[0] = 0; z4[1] = 0; z4[2] = 0; z4[3] = 0;
#if defined(__clang_major__) && (__clang_major__ >= 23)
  i32x8 z8;
#pragma unroll
  for (int i = 0; i < 8; ++i) z8[i] = 0;
  __builtin_amdgcn_tensor_load_to_lds(g0, g1, z4, z4, z8, 0);
#else
  __builtin_amdgcn_tensor_load_to_lds(g0, g1, z4, z4, 0);
#endif
  __builtin_amdgcn_s_wait_tensorcnt(0);
}

// ---------------------------------------------------------------------------
// Prep: fp32 [K,N] (optionally transposed source) -> f16 WMMA-B fragment
// layout; lane l -> n = nt*16+(l&15); half h -> k = kt*32 + 8*(l>>4) + (h&7) + 16*(h>>3).
__global__ void pack_wmma_b(const float* __restrict__ src, _Float16* __restrict__ dst,
                            int K, int N, int ktiles, int ntiles, int trans, int ldsrc) {
  int total = ktiles * ntiles * 512;
  for (int e = blockIdx.x * blockDim.x + threadIdx.x; e < total;
       e += gridDim.x * blockDim.x) {
    int f    = e >> 9;
    int r    = e & 511;
    int lane = r >> 4;
    int h    = r & 15;
    int kt   = f / ntiles;
    int nt   = f - kt * ntiles;
    int k    = kt * 32 + ((lane >> 4) << 3) + (h & 7) + ((h >> 3) << 4);
    int n    = nt * 16 + (lane & 15);
    float v = 0.0f;
    if (k < K && n < N)
      v = trans ? src[(size_t)n * ldsrc + k] : src[(size_t)k * ldsrc + n];
    dst[e] = (_Float16)v;
  }
}

__global__ void codebook_norms(const float* __restrict__ cb, float* __restrict__ cn) {
  int n = blockIdx.x * blockDim.x + threadIdx.x;
  if (n < NUM_EMB) {
    float s = 0.0f;
#pragma unroll 8
    for (int k = 0; k < LATENT; ++k) { float v = cb[n * LATENT + k]; s += v * v; }
    cn[n] = s;
  }
}

// ---------------------------------------------------------------------------
__global__ __launch_bounds__(THREADS)
void vqvae_fused(const float* __restrict__ x,
                 const float* __restrict__ enc_b1, const float* __restrict__ enc_b2,
                 const float* __restrict__ codebook,
                 const float* __restrict__ dec_b1, const float* __restrict__ dec_b2,
                 const float* __restrict__ cnorm,
                 const _Float16* __restrict__ w1p,  const _Float16* __restrict__ w2p,
                 const _Float16* __restrict__ cbp,  const _Float16* __restrict__ dw1p,
                 const _Float16* __restrict__ dw2p,
                 float* __restrict__ out_xr, float* __restrict__ out_z,
                 float* __restrict__ out_zq) {
  extern __shared__ char smem[];
  _Float16* sA    = reinterpret_cast<_Float16*>(smem + SA_OFF);
  float*    sStg  = reinterpret_cast<float*>(smem + STAGE_OFF);
  _Float16* sH    = reinterpret_cast<_Float16*>(smem + SH_OFF);
  _Float16* sZ    = reinterpret_cast<_Float16*>(smem + SZ_OFF);
  _Float16* sZq   = reinterpret_cast<_Float16*>(smem + SZQ_OFF);
  float*    sRedD = reinterpret_cast<float*>(smem + REDD_OFF);
  int*      sRedI = reinterpret_cast<int*>(smem + REDI_OFF);
  int*      sIdx  = reinterpret_cast<int*>(smem + IDX_OFF);

  const int tid  = threadIdx.x;
  const int lane = tid & 31;
  const int wid  = tid >> 5;
  const int m0   = (wid & 3) << 4;   // wave's 16-row tile
  const int wn   = wid >> 2;         // wave's N half (0/1)
  const int l15  = lane & 15;
  const int rhi  = (lane >> 4) << 3; // +0/+8 C/D row offset
  const int row0 = blockIdx.x * TILE_M;

  // ---- TDM: one wave DMAs the whole 64x514 fp32 x-tile into LDS -----------
  if (wid == 0) {
    unsigned ldsOff =
        (unsigned)(uintptr_t)(smem + STAGE_OFF);   // flat LDS addr low bits
    tdm_load_xtile(x, row0, ldsOff);
  }
  // prefetch next block's x tile while TDM runs -> global_prefetch_b8
  if (blockIdx.x + 1 < gridDim.x)
    __builtin_prefetch(x + (size_t)(row0 + TILE_M) * FEATURE + tid * 128, 0, 0);
  __syncthreads();

  // ---- convert staged fp32 -> f16 sA, zero-padded to K=544 ----------------
  for (int e = tid; e < TILE_M * KP1; e += THREADS) {
    int r = e / KP1, c = e - r * KP1;
    float v = (c < FEATURE) ? sStg[r * FEATURE + c] : 0.0f;
    sA[r * SA_STRIDE + c] = (_Float16)v;
  }
  __syncthreads();

  // ---- GEMM1: h = ELU(x @ W1 + b1)   [64x544]x[544x512] -------------------
  {
    v8f acc[16];
#pragma unroll
    for (int j = 0; j < 16; ++j) acc[j] = vzero();
    gemm_tile<16>(sA, SA_STRIDE, m0, w1p, 17, 32, wn * 16, lane, acc);
#pragma unroll
    for (int j = 0; j < 16; ++j) {
      int n = (wn * 16 + j) * 16 + l15;
      float b = enc_b1[n];
#pragma unroll
      for (int i = 0; i < 8; ++i) {
        int r = m0 + i + rhi;
        sH[r * SH_STRIDE + n] = (_Float16)elu_f(acc[j][i] + b);
      }
    }
  }
  __syncthreads();

  // ---- GEMM2: z = h @ W2 + b2   [64x512]x[512x128] ------------------------
  {
    v8f acc[4];
#pragma unroll
    for (int j = 0; j < 4; ++j) acc[j] = vzero();
    gemm_tile<4>(sH, SH_STRIDE, m0, w2p, 16, 8, wn * 4, lane, acc);
#pragma unroll
    for (int j = 0; j < 4; ++j) {
      int n = (wn * 4 + j) * 16 + l15;
      float b = enc_b2[n];
#pragma unroll
      for (int i = 0; i < 8; ++i) {
        int r = m0 + i + rhi;
        float zv = acc[j][i] + b;
        out_z[(size_t)(row0 + r) * LATENT + n] = zv;
        sZ[r * SZ_STRIDE + n] = (_Float16)zv;
      }
    }
  }
  __syncthreads();

  // ---- distances: argmin_n (||c_n||^2 - 2 z.c_n); ||z||^2 drops out -------
  {
    v8f acc[16];
#pragma unroll
    for (int j = 0; j < 16; ++j) acc[j] = vzero();
    gemm_tile<16>(sZ, SZ_STRIDE, m0, cbp, 4, 32, wn * 16, lane, acc);

    float bd[8]; int bi[8];
#pragma unroll
    for (int i = 0; i < 8; ++i) { bd[i] = 3.4e38f; bi[i] = 0; }
#pragma unroll
    for (int j = 0; j < 16; ++j) {
      int n = (wn * 16 + j) * 16 + l15;
      float cn = cnorm[n];
#pragma unroll
      for (int i = 0; i < 8; ++i) {
        float d = cn - 2.0f * acc[j][i];
        if (d < bd[i] || (d == bd[i] && n < bi[i])) { bd[i] = d; bi[i] = n; }
      }
    }
#pragma unroll
    for (int m = 8; m >= 1; m >>= 1) {          // 16-lane argmin tree (wave32)
#pragma unroll
      for (int i = 0; i < 8; ++i) {
        float od = __shfl_xor(bd[i], m, 32);
        int   oi = __shfl_xor(bi[i], m, 32);
        if (od < bd[i] || (od == bd[i] && oi < bi[i])) { bd[i] = od; bi[i] = oi; }
      }
    }
    if (l15 == 0) {
#pragma unroll
      for (int i = 0; i < 8; ++i) {
        int r = m0 + i + rhi;
        sRedD[r * 2 + wn] = bd[i];
        sRedI[r * 2 + wn] = bi[i];
      }
    }
  }
  __syncthreads();

  if (tid < TILE_M) {
    float d0 = sRedD[tid * 2 + 0], d1 = sRedD[tid * 2 + 1];
    int   i0 = sRedI[tid * 2 + 0], i1 = sRedI[tid * 2 + 1];
    sIdx[tid] = (d1 < d0 || (d1 == d0 && i1 < i0)) ? i1 : i0;
  }
  __syncthreads();

  // ---- gather z_q = codebook[idx]; emit fp32 + stage f16 ------------------
  for (int e = tid; e < TILE_M * LATENT; e += THREADS) {
    int r = e >> 7, c = e & 127;
    float v = codebook[(size_t)sIdx[r] * LATENT + c];
    out_zq[(size_t)(row0 + r) * LATENT + c] = v;
    sZq[r * SZ_STRIDE + c] = (_Float16)v;
  }
  __syncthreads();

  // ---- GEMM3: d = ELU(z_q @ dW1 + db1)   [64x128]x[128x512] ---------------
  {
    v8f acc[16];
#pragma unroll
    for (int j = 0; j < 16; ++j) acc[j] = vzero();
    gemm_tile<16>(sZq, SZ_STRIDE, m0, dw1p, 4, 32, wn * 16, lane, acc);
#pragma unroll
    for (int j = 0; j < 16; ++j) {
      int n = (wn * 16 + j) * 16 + l15;
      float b = dec_b1[n];
#pragma unroll
      for (int i = 0; i < 8; ++i) {
        int r = m0 + i + rhi;
        sH[r * SH_STRIDE + n] = (_Float16)elu_f(acc[j][i] + b);
      }
    }
  }
  __syncthreads();

  // ---- GEMM4: x_recon = sigmoid(d @ dW2 + db2)  [64x512]x[512x528pad] -----
  // n-half 0 -> n-tiles 0..16 (17 accs), n-half 1 -> 17..32 (16 accs):
  // A fragments loaded once per k-tile, accumulators stay resident.
  if (wn == 0) {
    v8f acc[17];
#pragma unroll
    for (int j = 0; j < 17; ++j) acc[j] = vzero();
    gemm_tile<17>(sH, SH_STRIDE, m0, dw2p, 16, NT4, 0, lane, acc);
    gemm4_epilogue<17>(acc, 0, l15, rhi, m0, row0, dec_b2, out_xr);
  } else {
    v8f acc[16];
#pragma unroll
    for (int j = 0; j < 16; ++j) acc[j] = vzero();
    gemm_tile<16>(sH, SH_STRIDE, m0, dw2p, 16, NT4, 17, lane, acc);
    gemm4_epilogue<16>(acc, 17, l15, rhi, m0, row0, dec_b2, out_xr);
  }
}

// ---------------------------------------------------------------------------
extern "C" void kernel_launch(void* const* d_in, const int* in_sizes, int n_in,
                              void* d_out, int out_size, void* d_ws, size_t ws_size,
                              hipStream_t stream) {
  const float* x      = (const float*)d_in[0];
  const float* enc_w1 = (const float*)d_in[1];
  const float* enc_b1 = (const float*)d_in[2];
  const float* enc_w2 = (const float*)d_in[3];
  const float* enc_b2 = (const float*)d_in[4];
  const float* cb     = (const float*)d_in[5];
  const float* dec_w1 = (const float*)d_in[6];
  const float* dec_b1 = (const float*)d_in[7];
  const float* dec_w2 = (const float*)d_in[8];
  const float* dec_b2 = (const float*)d_in[9];

  // workspace: f16 fragment-packed weights + codebook norms (~1.5 MB)
  _Float16* w1p  = (_Float16*)d_ws;           // 17*32*512
  _Float16* w2p  = w1p  + 17 * 32 * 512;      // 16*8*512
  _Float16* cbp  = w2p  + 16 * 8  * 512;      //  4*32*512
  _Float16* dw1p = cbp  +  4 * 32 * 512;      //  4*32*512
  _Float16* dw2p = dw1p +  4 * 32 * 512;      // 16*33*512
  float*    cn   = (float*)(dw2p + 16 * 33 * 512);

  pack_wmma_b<<<(17 * 32 * 512 + 255) / 256, 256, 0, stream>>>(
      enc_w1, w1p, FEATURE, HID, 17, 32, 0, HID);
  pack_wmma_b<<<(16 * 8 * 512 + 255) / 256, 256, 0, stream>>>(
      enc_w2, w2p, HID, LATENT, 16, 8, 0, LATENT);
  pack_wmma_b<<<(4 * 32 * 512 + 255) / 256, 256, 0, stream>>>(
      cb, cbp, LATENT, NUM_EMB, 4, 32, /*trans=*/1, LATENT);   // B = codebook^T
  pack_wmma_b<<<(4 * 32 * 512 + 255) / 256, 256, 0, stream>>>(
      dec_w1, dw1p, LATENT, HID, 4, 32, 0, HID);
  pack_wmma_b<<<(16 * 33 * 512 + 255) / 256, 256, 0, stream>>>(
      dec_w2, dw2p, HID, FEATURE, 16, 33, 0, FEATURE);
  codebook_norms<<<2, 256, 0, stream>>>(cb, cn);

  float* out_xr = (float*)d_out;
  float* out_z  = out_xr + (size_t)B_ROWS * FEATURE;
  float* out_zq = out_z  + (size_t)B_ROWS * LATENT;

  // ~204 KB dynamic LDS per workgroup (320 KB WGP on CDNA5)
  (void)hipFuncSetAttribute(reinterpret_cast<const void*>(vqvae_fused),
                            hipFuncAttributeMaxDynamicSharedMemorySize, SMEM_BYTES);

  vqvae_fused<<<B_ROWS / TILE_M, THREADS, SMEM_BYTES, stream>>>(
      x, enc_b1, enc_b2, cb, dec_b1, dec_b2, cn,
      w1p, w2p, cbp, dw1p, dw2p, out_xr, out_z, out_zq);
}
